// Model_35905926595230
// MI455X (gfx1250) — compile-verified, matrix operand order checked by reading, not run on hardware
//
#include <hip/hip_runtime.h>
#include <stdint.h>

// ---------------------------------------------------------------------------
// RETRO-style transformer forward for gfx1250 (MI455X, wave32, WMMA bf16).
// All matmuls use v_wmma_f32_16x16x32_bf16; tiles staged into LDS with
// GLOBAL_LOAD_ASYNC_TO_LDS_B128 (ASYNCcnt) and double-buffered.
// ---------------------------------------------------------------------------

typedef __attribute__((ext_vector_type(16))) __bf16 v16bf;
typedef __attribute__((ext_vector_type(8)))  float  v8f;

union FragU { unsigned u[8]; v16bf v; };

namespace {
constexpr int Dm    = 768;
constexpr int Hh    = 12;
constexpr int DKk   = 64;
constexpr int DFFf  = 3072;
constexpr int Bb    = 2;
constexpr int SEQn  = 1024;
constexpr int VOCAB = 32000;
constexpr float SCALE = 0.125f;  // 1/sqrt(64)
}

__device__ __forceinline__ unsigned short f2bf(float f) {
  union { float f; unsigned u; } c; c.f = f;
  unsigned u = c.u;
  return (unsigned short)((u + 0x7FFFu + ((u >> 16) & 1u)) >> 16);
}

// CDNA5 async global->LDS copy (tracked by ASYNCcnt, no VGPR staging).
__device__ __forceinline__ void async_ld_b128(unsigned lds_off, const void* gaddr) {
  asm volatile("global_load_async_to_lds_b128 %0, %1, off"
               :: "v"(lds_off), "v"(gaddr) : "memory");
}

// ------------------------------- GEMM --------------------------------------
// C[z] = alpha * A[z] @ B[z] + bias + add ; optional relu; f32 and/or bf16 out.
// Batch z = outer*inner + in; B-side outer index divided by bdiv.
// A: [M,K] row-major bf16. B: [K,N] row-major (btrans=0) or [N,K] (btrans=1).
struct GemmP {
  const unsigned short* A; const unsigned short* B;
  float* C; unsigned short* Cbf;
  const float* bias; const float* add;
  long long lda, ldb, ldc, lad;
  long long Ab0, Ab1, Bb0, Bb1, Cb0, Cb1;
  int M, N, K;
  int inner, batch, btrans, bdiv, relu;
  float alpha;
};

#define BM 128
#define BN 128
#define BK 32
#define ASTR 40
#define BSTR 40

__global__ __launch_bounds__(256) void gemm_bf16_kernel(GemmP p) {
  __shared__ __align__(16) unsigned short As[2][BM * ASTR];
  __shared__ __align__(16) unsigned short Bs[2][BN * BSTR];

  const int tid   = threadIdx.x;
  const int z     = blockIdx.z;
  const int in_i  = z % p.inner;
  const int outer = z / p.inner;

  const unsigned short* Ag = p.A + (long long)outer * p.Ab0 + (long long)in_i * p.Ab1;
  const unsigned short* Bg = p.B + (long long)(outer / p.bdiv) * p.Bb0 + (long long)in_i * p.Bb1;
  const long long coff = (long long)outer * p.Cb0 + (long long)in_i * p.Cb1;

  const int m0 = blockIdx.y * BM;
  const int n0 = blockIdx.x * BN;

  const int lane = tid & 31;
  const int w    = tid >> 5;
  const int wm   = w & 3;     // 4 waves over M (32 rows each)
  const int wn   = w >> 2;    // 2 waves over N (64 cols each)
  const int m16  = lane & 15;
  const int kh   = lane >> 4;

  const unsigned asBase = (unsigned)(uintptr_t)(&As[0][0]);
  const unsigned bsBase = (unsigned)(uintptr_t)(&Bs[0][0]);

  int kofs[8];
#pragma unroll
  for (int j = 0; j < 8; ++j) {
    int base = (j < 4) ? (j * 2) : (16 + (j - 4) * 2);
    kofs[j] = base + kh * 8;  // ISA 16-bit A/B fragment K mapping
  }

  v8f acc[2][4];
#pragma unroll
  for (int mi = 0; mi < 2; ++mi)
#pragma unroll
    for (int ni = 0; ni < 4; ++ni)
      acc[mi][ni] = (v8f){0.f, 0.f, 0.f, 0.f, 0.f, 0.f, 0.f, 0.f};

  // OOB rows are clamped (not predicated): garbage lands only in rows/cols
  // that the epilogue masks; EXEC stays all-ones as WMMA requires.
  auto stageA = [&](int kt, int buf) {
    const int k0 = kt * BK;
#pragma unroll
    for (int it = 0; it < 2; ++it) {
      int c   = tid + it * 256;
      int row = c >> 2;
      int kc  = (c & 3) << 3;
      int gm  = m0 + row; if (gm > p.M - 1) gm = p.M - 1;
      const unsigned short* ga = Ag + (long long)gm * p.lda + k0 + kc;
      unsigned lo = asBase + (unsigned)((buf * (BM * ASTR) + row * ASTR + kc) * 2);
      async_ld_b128(lo, ga);
    }
  };
  auto stageB = [&](int kt, int buf) {
    const int k0 = kt * BK;
    if (p.btrans) {
#pragma unroll
      for (int it = 0; it < 2; ++it) {
        int c   = tid + it * 256;
        int row = c >> 2;      // n
        int kc  = (c & 3) << 3;
        int gn  = n0 + row; if (gn > p.N - 1) gn = p.N - 1;
        const unsigned short* gb = Bg + (long long)gn * p.ldb + k0 + kc;
        unsigned lo = bsBase + (unsigned)((buf * (BN * BSTR) + row * BSTR + kc) * 2);
        async_ld_b128(lo, gb);
      }
    } else {
      unsigned short* bs = &Bs[buf][0];
#pragma unroll
      for (int it = 0; it < 2; ++it) {
        int c  = tid + it * 256;
        int k  = c >> 4;             // 0..31
        int nc = (c & 15) << 3;      // 0..120
        unsigned v0 = 0, v1 = 0, v2 = 0, v3 = 0;
        if (n0 + nc < p.N) {
          uint4 val = *reinterpret_cast<const uint4*>(Bg + (long long)(k0 + k) * p.ldb + n0 + nc);
          v0 = val.x; v1 = val.y; v2 = val.z; v3 = val.w;
        }
        bs[(nc + 0) * BSTR + k] = (unsigned short)(v0);
        bs[(nc + 1) * BSTR + k] = (unsigned short)(v0 >> 16);
        bs[(nc + 2) * BSTR + k] = (unsigned short)(v1);
        bs[(nc + 3) * BSTR + k] = (unsigned short)(v1 >> 16);
        bs[(nc + 4) * BSTR + k] = (unsigned short)(v2);
        bs[(nc + 5) * BSTR + k] = (unsigned short)(v2 >> 16);
        bs[(nc + 6) * BSTR + k] = (unsigned short)(v3);
        bs[(nc + 7) * BSTR + k] = (unsigned short)(v3 >> 16);
      }
    }
  };

  const int kTiles = p.K / BK;
  stageA(0, 0);
  stageB(0, 0);

  for (int kt = 0; kt < kTiles; ++kt) {
    const int cur = kt & 1;
    const bool hasNext = (kt + 1) < kTiles;
    if (hasNext) {            // prefetch next tile into the other buffer
      stageA(kt + 1, cur ^ 1);
      stageB(kt + 1, cur ^ 1);
      // in-order ASYNCcnt: leaving next tile's loads in flight guarantees
      // the current tile's loads have landed.
      if (p.btrans) asm volatile("s_wait_asynccnt 0x4" ::: "memory");
      else          asm volatile("s_wait_asynccnt 0x2" ::: "memory");
    } else {
      asm volatile("s_wait_asynccnt 0x0" ::: "memory");
    }
    __syncthreads();

    FragU a[2], b[4];
#pragma unroll
    for (int mi = 0; mi < 2; ++mi) {
      int r = wm * 32 + mi * 16 + m16;
#pragma unroll
      for (int j = 0; j < 8; ++j)
        a[mi].u[j] = *reinterpret_cast<const unsigned*>(&As[cur][r * ASTR + kofs[j]]);
    }
#pragma unroll
    for (int ni = 0; ni < 4; ++ni) {
      int cidx = wn * 64 + ni * 16 + m16;
#pragma unroll
      for (int j = 0; j < 8; ++j)
        b[ni].u[j] = *reinterpret_cast<const unsigned*>(&Bs[cur][cidx * BSTR + kofs[j]]);
    }
#pragma unroll
    for (int mi = 0; mi < 2; ++mi)
#pragma unroll
      for (int ni = 0; ni < 4; ++ni)
        acc[mi][ni] = __builtin_amdgcn_wmma_f32_16x16x32_bf16(
            false, a[mi].v, false, b[ni].v, (short)0, acc[mi][ni], false, false);
    __syncthreads();
  }

  // ---- epilogue: scale + bias + residual + relu, f32/bf16 stores ----
#pragma unroll
  for (int mi = 0; mi < 2; ++mi) {
    int mbase = m0 + wm * 32 + mi * 16 + kh * 8;
#pragma unroll
    for (int ni = 0; ni < 4; ++ni) {
      int nn = n0 + wn * 64 + ni * 16 + m16;
      if (nn >= p.N) continue;
      float bv = p.bias ? p.bias[nn] : 0.f;
#pragma unroll
      for (int r = 0; r < 8; ++r) {
        int mm = mbase + r;
        if (mm >= p.M) continue;
        float v = acc[mi][ni][r] * p.alpha + bv;
        if (p.add)  v += p.add[(long long)mm * p.lad + nn];
        if (p.relu) v = fmaxf(v, 0.f);
        long long idx = coff + (long long)mm * p.ldc + nn;
        if (p.C)   p.C[idx]   = v;
        if (p.Cbf) p.Cbf[idx] = f2bf(v);
      }
    }
  }
}

// --------------------------- support kernels --------------------------------

__global__ void cvt_kernel(const float* __restrict__ in, unsigned short* __restrict__ out,
                           long long n) {
  long long stride = (long long)gridDim.x * 256;
  for (long long i = (long long)blockIdx.x * 256 + threadIdx.x; i < n; i += stride)
    out[i] = f2bf(in[i]);
}

// f32 [K,N] -> bf16 [N,K] (weights pre-transposed so GEMMs use the async path)
__global__ void cvt_t_kernel(const float* __restrict__ in, unsigned short* __restrict__ out,
                             int K, int N) {
  long long total = (long long)K * N;
  long long stride = (long long)gridDim.x * 256;
  for (long long i = (long long)blockIdx.x * 256 + threadIdx.x; i < total; i += stride) {
    long long k = i / N, n = i % N;
    out[n * (long long)K + k] = f2bf(in[i]);
  }
}

__global__ void embed_kernel(const int* __restrict__ ids, const float* __restrict__ emb,
                             float* __restrict__ out) {
  int t = blockIdx.x;
  const float* src = emb + (long long)ids[t] * Dm;
  float* dst = out + (long long)t * Dm;
  for (int j = threadIdx.x; j < Dm; j += 256) dst[j] = src[j];
}

// LayerNorm over D=768; row r -> (b = r/seqOut, t = r%seqOut, src = t+shift).
// Rows past seqIn emit zeros (CCA right-pad after the causal shift).
__global__ __launch_bounds__(256) void ln_kernel(const float* __restrict__ X,
                                                 const float* __restrict__ g,
                                                 const float* __restrict__ b,
                                                 unsigned short* __restrict__ Y,
                                                 int seqOut, int seqIn, int shift) {
  int row = blockIdx.x;
  int bi  = row / seqOut;
  int t   = row % seqOut;
  int s   = t + shift;
  unsigned short* y = Y + (long long)row * Dm;
  int tid = threadIdx.x;
  if (s >= seqIn) {
    for (int j = tid; j < Dm; j += 256) y[j] = 0;
    return;
  }
  const float* x = X + ((long long)bi * seqIn + s) * Dm;
  __shared__ float red[256];
  float acc = 0.f;
  for (int j = tid; j < Dm; j += 256) acc += x[j];
  red[tid] = acc; __syncthreads();
  for (int off = 128; off > 0; off >>= 1) {
    if (tid < off) red[tid] += red[tid + off];
    __syncthreads();
  }
  float mean = red[0] * (1.f / Dm);
  __syncthreads();
  acc = 0.f;
  for (int j = tid; j < Dm; j += 256) { float d = x[j] - mean; acc += d * d; }
  red[tid] = acc; __syncthreads();
  for (int off = 128; off > 0; off >>= 1) {
    if (tid < off) red[tid] += red[tid + off];
    __syncthreads();
  }
  float inv = rsqrtf(red[0] * (1.f / Dm) + 1e-5f);
  for (int j = tid; j < Dm; j += 256)
    y[j] = f2bf((x[j] - mean) * inv * g[j] + b[j]);
}

// Row softmax (scores already scaled). causal: valid = (row%seq)+1 else L.
__global__ __launch_bounds__(256) void softmax_kernel(const float* __restrict__ S,
                                                      unsigned short* __restrict__ P,
                                                      int L, int seq, int causal) {
  int row = blockIdx.x;
  int valid = causal ? ((row % seq) + 1) : L;
  const float* s = S + (long long)row * L;
  unsigned short* pr = P + (long long)row * L;
  int tid = threadIdx.x;
  __shared__ float red[256];
  float mx = -1e30f;
  for (int j = tid; j < valid; j += 256) mx = fmaxf(mx, s[j]);
  red[tid] = mx; __syncthreads();
  for (int off = 128; off > 0; off >>= 1) {
    if (tid < off) red[tid] = fmaxf(red[tid], red[tid + off]);
    __syncthreads();
  }
  mx = red[0]; __syncthreads();
  float sum = 0.f;
  for (int j = tid; j < valid; j += 256) sum += __expf(s[j] - mx);
  red[tid] = sum; __syncthreads();
  for (int off = 128; off > 0; off >>= 1) {
    if (tid < off) red[tid] += red[tid + off];
    __syncthreads();
  }
  float inv = 1.f / red[0];
  for (int j = tid; j < L; j += 256)
    pr[j] = (j < valid) ? f2bf(__expf(s[j] - mx) * inv) : (unsigned short)0;
}

// h[b,t,:] += (t>=63) ? o[b,t-63,:] : 0   (chunked-CA causal shift residual)
__global__ void ccadd_kernel(float* __restrict__ h, const float* __restrict__ o) {
  const long long total = (long long)Bb * SEQn * Dm;
  long long stride = (long long)gridDim.x * 256;
  for (long long i = (long long)blockIdx.x * 256 + threadIdx.x; i < total; i += stride) {
    long long d = i % Dm;
    long long t = (i / Dm) % SEQn;
    long long b = i / ((long long)Dm * SEQn);
    if (t >= 63) h[i] += o[(b * SEQn + (t - 63)) * Dm + d];
  }
}

// ------------------------------- host side ----------------------------------

namespace {

struct Arena {
  char* base; size_t off;
  void* alloc(size_t bytes) {
    void* p = base + off;
    off += (bytes + 255) & ~(size_t)255;
    return p;
  }
};

struct AttnW {  // weights stored transposed [N,K] bf16
  const unsigned short *qw, *kw, *vw, *ow;
  const float *qb, *kb, *vb, *ob, *ng, *nb;
};
struct FfwW {
  const unsigned short *w1, *w2;  // transposed
  const float *b1, *b2, *ng, *nb;
};
struct Scratch {
  unsigned short *qbf, *kbf, *vbf, *probs, *ctxbf, *lnb, *mid;
  float* scores;
};

const unsigned short* cvw_t(hipStream_t st, Arena& ar, const float* w, int K, int N) {
  unsigned short* d = (unsigned short*)ar.alloc((size_t)K * N * 2);
  long long n = (long long)K * N;
  long long blocks = (n + 255) / 256; if (blocks > 4096) blocks = 4096;
  cvt_t_kernel<<<dim3((unsigned)blocks), dim3(256), 0, st>>>(w, d, K, N);
  return d;
}

void launch_gemm(hipStream_t st, const GemmP& p) {
  dim3 grid((p.N + BN - 1) / BN, (p.M + BM - 1) / BM, (unsigned)p.batch);
  gemm_bf16_kernel<<<grid, dim3(256), 0, st>>>(p);
}

// A [M,K] bf16 @ Wt (transposed [N,K] bf16) + bias (+add, relu)
void lin(hipStream_t st, const unsigned short* A, int M, const unsigned short* Wt,
         int K, int N, const float* bias, float* C, unsigned short* Cbf,
         const float* add, int relu) {
  GemmP p{};
  p.A = A; p.lda = K; p.B = Wt; p.ldb = K; p.btrans = 1; p.bdiv = 1;
  p.C = C; p.Cbf = Cbf; p.ldc = N; p.lad = N;
  p.bias = bias; p.add = add;
  p.M = M; p.N = N; p.K = K;
  p.inner = 1; p.batch = 1; p.alpha = 1.f; p.relu = relu;
  launch_gemm(st, p);
}

void ln_rows(hipStream_t st, const float* X, int rows, int seqOut, int seqIn, int shift,
             const float* g, const float* b, unsigned short* Y) {
  ln_kernel<<<dim3((unsigned)rows), dim3(256), 0, st>>>(X, g, b, Y, seqOut, seqIn, shift);
}

// Multi-head attention: qin [qOuter*Mq, 768] bf16 (pre-normalized),
// kvin [kvOuter*Nk, 768] bf16. outC/resid f32 [qTok, 768].
void mha(hipStream_t st, Scratch& s, const unsigned short* qin, int qOuter, int Mq,
         const unsigned short* kvin, int kvOuter, int Nk, const AttnW& w, int causal,
         float* outC, const float* resid) {
  const int qTok = qOuter * Mq, kvTok = kvOuter * Nk;
  lin(st, qin,  qTok,  w.qw, Dm, Dm, w.qb, nullptr, s.qbf, nullptr, 0);
  lin(st, kvin, kvTok, w.kw, Dm, Dm, w.kb, nullptr, s.kbf, nullptr, 0);
  lin(st, kvin, kvTok, w.vw, Dm, Dm, w.vb, nullptr, s.vbf, nullptr, 0);
  const int bdiv = qOuter / kvOuter;

  GemmP p{};  // scores = SCALE * Q @ K^T  per (outer, head)
  p.A = s.qbf; p.lda = Dm; p.Ab0 = (long long)Mq * Dm; p.Ab1 = DKk;
  p.B = s.kbf; p.ldb = Dm; p.Bb0 = (long long)Nk * Dm; p.Bb1 = DKk;
  p.btrans = 1; p.bdiv = bdiv;
  p.C = s.scores; p.ldc = Nk;
  p.Cb0 = (long long)Hh * Mq * Nk; p.Cb1 = (long long)Mq * Nk;
  p.M = Mq; p.N = Nk; p.K = DKk;
  p.inner = Hh; p.batch = qOuter * Hh; p.alpha = SCALE;
  launch_gemm(st, p);

  softmax_kernel<<<dim3((unsigned)(qOuter * Hh * Mq)), dim3(256), 0, st>>>(
      s.scores, s.probs, Nk, Mq, causal);

  GemmP c{};  // ctx = P @ V, written as [tok, h*64+d] bf16
  c.A = s.probs; c.lda = Nk; c.Ab0 = (long long)Hh * Mq * Nk; c.Ab1 = (long long)Mq * Nk;
  c.B = s.vbf; c.ldb = Dm; c.Bb0 = (long long)Nk * Dm; c.Bb1 = DKk;
  c.btrans = 0; c.bdiv = bdiv;
  c.Cbf = s.ctxbf; c.ldc = Dm; c.Cb0 = (long long)Mq * Dm; c.Cb1 = DKk;
  c.M = Mq; c.N = DKk; c.K = Nk;
  c.inner = Hh; c.batch = qOuter * Hh; c.alpha = 1.f;
  launch_gemm(st, c);

  lin(st, s.ctxbf, qTok, w.ow, Dm, Dm, w.ob, outC, nullptr, resid, 0);
}

void ffw_block(hipStream_t st, float* hbuf, int tok, const FfwW& w, Scratch& s) {
  ln_rows(st, hbuf, tok, tok, tok, 0, w.ng, w.nb, s.lnb);
  lin(st, s.lnb, tok, w.w1, Dm, DFFf, w.b1, nullptr, s.mid, nullptr, 1);
  lin(st, s.mid, tok, w.w2, DFFf, Dm, w.b2, hbuf, nullptr, hbuf, 0);
}

// Leaf order (jax tree: dict keys sorted alphabetically, lists in order).
AttnW load_attn(hipStream_t st, Arena& ar, void* const* din, int base) {
  AttnW w;
  w.kb = (const float*)din[base + 0];
  const float* kw = (const float*)din[base + 1];
  w.nb = (const float*)din[base + 2];
  w.ng = (const float*)din[base + 3];
  w.ob = (const float*)din[base + 4];
  const float* ow = (const float*)din[base + 5];
  w.qb = (const float*)din[base + 6];
  const float* qw = (const float*)din[base + 7];
  w.vb = (const float*)din[base + 8];
  const float* vw = (const float*)din[base + 9];
  w.qw = cvw_t(st, ar, qw, Dm, Dm);
  w.kw = cvw_t(st, ar, kw, Dm, Dm);
  w.vw = cvw_t(st, ar, vw, Dm, Dm);
  w.ow = cvw_t(st, ar, ow, Dm, Dm);
  return w;
}

FfwW load_ffw(hipStream_t st, Arena& ar, void* const* din, int base) {
  FfwW w;
  w.b1 = (const float*)din[base + 0];
  w.b2 = (const float*)din[base + 1];
  w.nb = (const float*)din[base + 2];
  w.ng = (const float*)din[base + 3];
  const float* w1 = (const float*)din[base + 4];
  const float* w2 = (const float*)din[base + 5];
  w.w1 = cvw_t(st, ar, w1, Dm, DFFf);
  w.w2 = cvw_t(st, ar, w2, DFFf, Dm);
  return w;
}

}  // namespace

extern "C" void kernel_launch(void* const* d_in, const int* in_sizes, int n_in,
                              void* d_out, int out_size, void* d_ws, size_t ws_size,
                              hipStream_t stream) {
  (void)in_sizes; (void)n_in; (void)out_size; (void)ws_size;
  Arena ar{(char*)d_ws, 0};

  // --- weights (bf16 transposed conversions into workspace) ---
  AttnW attnW[4], ccaW[2], encAttnW[2], encCaW;
  FfwW  ffwW[4], encFfwW[2];
  for (int p = 0; p < 4; ++p) attnW[p]   = load_attn(stream, ar, d_in, 0 + 10 * p);
  for (int i = 0; i < 2; ++i) ccaW[i]    = load_attn(stream, ar, d_in, 40 + 10 * i);
  const float* emb = (const float*)d_in[60];
  for (int q = 0; q < 2; ++q) encAttnW[q] = load_attn(stream, ar, d_in, 61 + 10 * q);
  encCaW = load_attn(stream, ar, d_in, 81);
  for (int q = 0; q < 2; ++q) encFfwW[q]  = load_ffw(stream, ar, d_in, 91 + 6 * q);
  const float* nh_b = (const float*)d_in[103];
  const float* nh_g = (const float*)d_in[104];
  for (int p = 0; p < 4; ++p) ffwW[p] = load_ffw(stream, ar, d_in, 105 + 6 * p);
  const float* ne_b = (const float*)d_in[129];
  const float* ne_g = (const float*)d_in[130];
  const float* rb   = (const float*)d_in[131];
  const unsigned short* rwT = cvw_t(stream, ar, (const float*)d_in[132], Dm, VOCAB);
  const int* ret = (const int*)d_in[133];
  const int* x   = (const int*)d_in[134];

  // --- activation buffers ---
  const int decTok = Bb * SEQn;                 // 2048
  const int encTok = Bb * 16 * 2 * 128;         // 8192
  float* h    = (float*)ar.alloc((size_t)decTok * Dm * 4);
  float* ebuf = (float*)ar.alloc((size_t)encTok * Dm * 4);
  float* ocaa = (float*)ar.alloc((size_t)decTok * Dm * 4);
  unsigned short* hsenc = (unsigned short*)ar.alloc((size_t)decTok * Dm * 2);
  unsigned short* eln   = (unsigned short*)ar.alloc((size_t)encTok * Dm * 2);
  unsigned short* hbf   = (unsigned short*)ar.alloc((size_t)decTok * Dm * 2);

  Scratch s;
  s.qbf    = (unsigned short*)ar.alloc((size_t)encTok * Dm * 2);
  s.kbf    = (unsigned short*)ar.alloc((size_t)encTok * Dm * 2);
  s.vbf    = (unsigned short*)ar.alloc((size_t)encTok * Dm * 2);
  s.lnb    = (unsigned short*)ar.alloc((size_t)encTok * Dm * 2);
  s.ctxbf  = (unsigned short*)ar.alloc((size_t)encTok * Dm * 2);
  s.mid    = (unsigned short*)ar.alloc((size_t)encTok * DFFf * 2);
  const long long maxScore = (long long)Bb * Hh * SEQn * SEQn;  // 25.2M
  s.scores = (float*)ar.alloc((size_t)maxScore * 4);
  s.probs  = (unsigned short*)ar.alloc((size_t)maxScore * 2);

  // --- embeddings ---
  embed_kernel<<<dim3(decTok), dim3(256), 0, stream>>>(x, emb, h);
  embed_kernel<<<dim3(encTok), dim3(256), 0, stream>>>(ret, emb, ebuf);

  int pc = 0;
  for (int p = 0; p < 4; ++p) {
    // self-attention (causal)
    ln_rows(stream, h, decTok, decTok, decTok, 0, attnW[p].ng, attnW[p].nb, s.lnb);
    mha(stream, s, s.lnb, Bb, SEQn, s.lnb, Bb, SEQn, attnW[p], 1, h, h);

    if (p == 2) {
      // encoder over retrieved neighbors
      ln_rows(stream, h, decTok, decTok, decTok, 0, nh_g, nh_b, hsenc);
      for (int q = 0; q < 2; ++q) {
        ln_rows(stream, ebuf, encTok, encTok, encTok, 0,
                encAttnW[q].ng, encAttnW[q].nb, s.lnb);
        mha(stream, s, s.lnb, 64, 128, s.lnb, 64, 128, encAttnW[q], 0, ebuf, ebuf);
        if (q == 1) {  // encoder cross-attn to decoder chunks (bdiv=2 over n)
          ln_rows(stream, ebuf, encTok, encTok, encTok, 0, encCaW.ng, encCaW.nb, s.lnb);
          mha(stream, s, s.lnb, 64, 128, hsenc, 32, 64, encCaW, 0, ebuf, ebuf);
        }
        ffw_block(stream, ebuf, encTok, encFfwW[q], s);
      }
      ln_rows(stream, ebuf, encTok, encTok, encTok, 0, ne_g, ne_b, eln);
    }
    if (p == 2 || p == 3) {
      // chunked cross-attention: shift-63 LN + zero pad, joint softmax over 256
      ln_rows(stream, h, decTok, SEQn, SEQn, 63, ccaW[pc].ng, ccaW[pc].nb, s.lnb);
      mha(stream, s, s.lnb, 32, 64, eln, 32, 256, ccaW[pc], 0, ocaa, nullptr);
      ccadd_kernel<<<dim3(4096), dim3(256), 0, stream>>>(h, ocaa);
      ++pc;
    }
    ffw_block(stream, h, decTok, ffwW[p], s);
  }

  // logits = h @ rw + rb  (2048 x 32000)
  cvt_kernel<<<dim3(4096), dim3(256), 0, stream>>>(h, hbf, (long long)decTok * Dm);
  lin(stream, hbf, decTok, rwT, Dm, VOCAB, rb, (float*)d_out, nullptr, nullptr, 0);
}
